// CrossAttention_63118839382400
// MI455X (gfx1250) — matrix operand order repaired and checked by reading, not compiled, not run on hardware
//
#include <hip/hip_runtime.h>
#include <hip/hip_bf16.h>

// ---------------- problem constants ----------------
#define BB 4
#define LQ 1024
#define LK 2048
#define DD 1024
#define HH 16
#define DH 64
#define EPS 1e-6f
#define SCALE 0.125f // DH^-0.5

typedef __attribute__((ext_vector_type(16))) __bf16 v16bf;
typedef __attribute__((ext_vector_type(8)))  float  v8f;

// ---- WMMA fragment loaders (layouts per CDNA5 ISA 7.12.2) ----
// A (16x32, MxK): lane<16 -> M=lane, elems 0..7 = K kb..kb+7, elems 8..15 = K kb+16..kb+23, kb=(lane>>4)*8
__device__ __forceinline__ v16bf load_a_frag(const __bf16* p) {
  union { uint4 u[2]; v16bf v; } t;
  t.u[0] = *reinterpret_cast<const uint4*>(p);
  t.u[1] = *reinterpret_cast<const uint4*>(p + 16);
  return t.v;
}
// B (32x16, KxN): lane -> N=lane&15, elems 0..7 = K kb..kb+7, elems 8..15 = K kb+16..kb+23, kb=(lane>>4)*8
__device__ __forceinline__ v16bf load_b_frag(const __bf16* p) {
  union { uint4 u[2]; v16bf v; } t;
  t.u[0] = *reinterpret_cast<const uint4*>(p);
  t.u[1] = *reinterpret_cast<const uint4*>(p + 16);
  return t.v;
}
__device__ __forceinline__ v8f wmma_bf16(v16bf a, v16bf b, v8f c) {
  return __builtin_amdgcn_wmma_f32_16x16x32_bf16(false, a, false, b, (short)0, c, false, false);
}

// ---------------- fp32 -> bf16 convert ----------------
__global__ void cvt_bf16_kernel(const float* __restrict__ in, __bf16* __restrict__ out, int n) {
  int i = blockIdx.x * blockDim.x + threadIdx.x;
  if (i < n) out[i] = (__bf16)in[i];
}

// ---------------- generic NT GEMM: C[M,N] = A[M,K] * B[N,K]^T ----------------
// block = 256 threads (8 waves, 2x4): block tile 64(M) x 128(N); wave tile 32x32
// (2 A-frags x 2 B-frags -> 4 wmma per 32-K step; both operands reused).
__global__ void gemm_nt_bf16(const __bf16* __restrict__ A, const __bf16* __restrict__ Bm,
                             float* __restrict__ C, int M, int N, int K) {
  int lane = threadIdx.x & 31, wid = threadIdx.x >> 5;
  int waveM = wid >> 2, waveN = wid & 3;
  int m0 = blockIdx.y * 64 + waveM * 32;
  int n0 = blockIdx.x * 128 + waveN * 32;
  int row = lane & 15, hi = lane >> 4;

  const __bf16* arow0 = A  + (size_t)(m0 + row) * K + hi * 8;
  const __bf16* arow1 = A  + (size_t)(m0 + 16 + row) * K + hi * 8;
  const __bf16* brow0 = Bm + (size_t)(n0 + row) * K + hi * 8;
  const __bf16* brow1 = Bm + (size_t)(n0 + 16 + row) * K + hi * 8;

  v8f c00 = {}, c01 = {}, c10 = {}, c11 = {};
  for (int k0 = 0; k0 < K; k0 += 32) {
    __builtin_prefetch(arow0 + k0 + 256, 0, 1);  // global_prefetch_b8
    __builtin_prefetch(brow0 + k0 + 256, 0, 1);
    v16bf a0 = load_a_frag(arow0 + k0);
    v16bf a1 = load_a_frag(arow1 + k0);
    v16bf b0 = load_b_frag(brow0 + k0);
    v16bf b1 = load_b_frag(brow1 + k0);
    c00 = wmma_bf16(a0, b0, c00);
    c01 = wmma_bf16(a0, b1, c01);
    c10 = wmma_bf16(a1, b0, c10);
    c11 = wmma_bf16(a1, b1, c11);
  }
  // C layout: VGPR r -> row = r + hi*8, col = lane&15
#pragma unroll
  for (int r = 0; r < 8; r++) {
    size_t orowA = (size_t)(m0 + r + hi * 8) * N + n0;
    size_t orowB = (size_t)(m0 + 16 + r + hi * 8) * N + n0;
    C[orowA + row]      = c00[r];
    C[orowA + 16 + row] = c01[r];
    C[orowB + row]      = c10[r];
    C[orowB + 16 + row] = c11[r];
  }
}

// ---------------- RMSNorm + RoPE for Q (scale folded in) ----------------
__global__ void norm_rope_q_kernel(const float* __restrict__ Qp, const float* __restrict__ qw,
                                   const int* __restrict__ pos, __bf16* __restrict__ qhb) {
  __shared__ float sh[DH];
  __shared__ float wsum[2];
  int idx = blockIdx.x;
  int h = idx % HH, lq = (idx / HH) % LQ, b = idx / (HH * LQ);
  int t = threadIdx.x;

  float x = Qp[((size_t)b * LQ + lq) * DD + h * DH + t];
  float v = x * x;
#pragma unroll
  for (int m = 1; m < 32; m <<= 1) v += __shfl_xor(v, m);
  if ((t & 31) == 0) wsum[t >> 5] = v;
  __syncthreads();
  float total = wsum[0] + wsum[1];
  float xn = x * rsqrtf(total * (1.0f / DH) + EPS) * qw[t];
  sh[t] = xn;
  __syncthreads();
  if (t < 32) {
    float x1 = sh[t], x2 = sh[t + 32];
    float ang = (float)pos[lq] * __powf(10000.0f, -(float)t / 32.0f);
    float c = __cosf(ang), s = __sinf(ang);
    size_t ob = ((size_t)(b * HH + h) * LQ + lq) * DH;
    qhb[ob + t]      = (__bf16)((x1 * c - x2 * s) * SCALE);
    qhb[ob + t + 32] = (__bf16)((x1 * s + x2 * c) * SCALE);
  }
}

// ---------------- RMSNorm + RoPE for K; V copied bf16 TRANSPOSED [bh][DH][LK] ----------------
__global__ void norm_rope_kv_kernel(const float* __restrict__ KVp, const float* __restrict__ kw,
                                    const int* __restrict__ pos, __bf16* __restrict__ khb,
                                    __bf16* __restrict__ vhbT) {
  __shared__ float sh[DH];
  __shared__ float wsum[2];
  int idx = blockIdx.x;
  int h = idx % HH, lk = (idx / HH) % LK, b = idx / (HH * LK);
  int t = threadIdx.x;
  int bh = b * HH + h;

  size_t rbase = ((size_t)b * LK + lk) * (2 * DD) + h * (2 * DH);
  float xk = KVp[rbase + t];
  float xv = KVp[rbase + DH + t];

  float v = xk * xk;
#pragma unroll
  for (int m = 1; m < 32; m <<= 1) v += __shfl_xor(v, m);
  if ((t & 31) == 0) wsum[t >> 5] = v;
  __syncthreads();
  float total = wsum[0] + wsum[1];
  float xn = xk * rsqrtf(total * (1.0f / DH) + EPS) * kw[t];
  sh[t] = xn;

  // V transposed: [bh][d][lk] so attention B-fragments are K-contiguous
  vhbT[((size_t)bh * DH + t) * LK + lk] = (__bf16)xv;
  __syncthreads();
  if (t < 32) {
    float x1 = sh[t], x2 = sh[t + 32];
    float ang = (float)pos[lk] * __powf(10000.0f, -(float)t / 32.0f);
    float c = __cosf(ang), s = __sinf(ang);
    size_t ob = ((size_t)bh * LK + lk) * DH;
    khb[ob + t]      = (__bf16)(x1 * c - x2 * s);
    khb[ob + t + 32] = (__bf16)(x1 * s + x2 * c);
  }
}

// ---------------- flash attention: 1 wave per (b,h, 16-query tile) ----------------
// score tile 16x32 via 4 wmma (Kdim=DH=64); online softmax; P staged via wave-private
// LDS (C-layout -> A-layout); P*V via 4 wmma (Kdim=32, V read from transposed layout
// with two 16B per-lane loads). Output bf16 [b,lq,h*DH+dh].
__global__ void attn_kernel(const __bf16* __restrict__ qhb, const __bf16* __restrict__ khb,
                            const __bf16* __restrict__ vhbT, __bf16* __restrict__ ybf) {
  __shared__ __attribute__((aligned(64))) __bf16 pbuf[8][16][32];
  int lane = threadIdx.x & 31, wid = threadIdx.x >> 5;
  int w = blockIdx.x * 8 + wid;
  const int QTILES = LQ / 16;
  int bh = w / QTILES, qt = w % QTILES;
  int b = bh / HH, h = bh % HH;
  int row = lane & 15, hi = lane >> 4;
  int q0 = qt * 16;

  const __bf16* Q  = qhb  + (size_t)bh * LQ * DH;
  const __bf16* Kp = khb  + (size_t)bh * LK * DH;
  const __bf16* Vt = vhbT + (size_t)bh * DH * LK;   // [d][lk]

  // Q fragments (reused all iterations): d-chunks 0..31 and 32..63
  v16bf qa0 = load_a_frag(Q + (size_t)(q0 + row) * DH + 0  + hi * 8);
  v16bf qa1 = load_a_frag(Q + (size_t)(q0 + row) * DH + 32 + hi * 8);

  v8f acc0 = {}, acc1 = {}, acc2 = {}, acc3 = {};
  float mi[8], li[8];
#pragma unroll
  for (int r = 0; r < 8; r++) { mi[r] = -1e30f; li[r] = 0.0f; }

  for (int k0 = 0; k0 < LK; k0 += 32) {
    // --- scores: two 16x16 tiles covering k-positions [k0, k0+32) ---
    v8f s0 = {}, s1 = {};
    v16bf kb;
    kb = load_b_frag(Kp + (size_t)(k0 + row) * DH + 0  + hi * 8); s0 = wmma_bf16(qa0, kb, s0);
    kb = load_b_frag(Kp + (size_t)(k0 + row) * DH + 32 + hi * 8); s0 = wmma_bf16(qa1, kb, s0);
    kb = load_b_frag(Kp + (size_t)(k0 + 16 + row) * DH + 0  + hi * 8); s1 = wmma_bf16(qa0, kb, s1);
    kb = load_b_frag(Kp + (size_t)(k0 + 16 + row) * DH + 32 + hi * 8); s1 = wmma_bf16(qa1, kb, s1);

    // --- online softmax update (row stats per (r, half); xor masks stay in 16-lane half) ---
#pragma unroll
    for (int r = 0; r < 8; r++) {
      float t = fmaxf(s0[r], s1[r]);
      t = fmaxf(t, __shfl_xor(t, 1)); t = fmaxf(t, __shfl_xor(t, 2));
      t = fmaxf(t, __shfl_xor(t, 4)); t = fmaxf(t, __shfl_xor(t, 8));
      float mn = fmaxf(mi[r], t);
      float corr = __expf(mi[r] - mn);
      mi[r] = mn;
      float p0 = __expf(s0[r] - mn);
      float p1 = __expf(s1[r] - mn);
      float rs = p0 + p1;
      rs += __shfl_xor(rs, 1); rs += __shfl_xor(rs, 2);
      rs += __shfl_xor(rs, 4); rs += __shfl_xor(rs, 8);
      li[r] = li[r] * corr + rs;
      acc0[r] *= corr; acc1[r] *= corr; acc2[r] *= corr; acc3[r] *= corr;
      // stage P (C-layout -> row-major LDS): row = r+hi*8, cols lane&15 / +16
      pbuf[wid][r + hi * 8][row]      = (__bf16)p0;
      pbuf[wid][r + hi * 8][row + 16] = (__bf16)p1;
    }

    // --- P as A-fragment (16x32) ---
    v16bf pa = load_a_frag(&pbuf[wid][row][hi * 8]);

    // --- y += P * V : 4 d-tiles of 16; V B-frag = two 16B loads per lane ---
#pragma unroll
    for (int dt = 0; dt < 4; dt++) {
      v16bf vf = load_b_frag(Vt + (size_t)(dt * 16 + row) * LK + k0 + hi * 8);
      v8f* accp = (dt == 0) ? &acc0 : (dt == 1) ? &acc1 : (dt == 2) ? &acc2 : &acc3;
      *accp = wmma_bf16(pa, vf, *accp);
    }
  }

  // --- finalize: divide by l, write bf16 in [b, lq, h*DH + dh] ---
#pragma unroll
  for (int r = 0; r < 8; r++) {
    float inv = 1.0f / li[r];
    int orow = q0 + r + hi * 8;
    size_t base = ((size_t)b * LQ + orow) * DD + h * DH;
    ybf[base + 0  + row] = (__bf16)(acc0[r] * inv);
    ybf[base + 16 + row] = (__bf16)(acc1[r] * inv);
    ybf[base + 32 + row] = (__bf16)(acc2[r] * inv);
    ybf[base + 48 + row] = (__bf16)(acc3[r] * inv);
  }
}

// ---------------- host launcher ----------------
extern "C" void kernel_launch(void* const* d_in, const int* in_sizes, int n_in,
                              void* d_out, int out_size, void* d_ws, size_t ws_size,
                              hipStream_t stream) {
  const float* queries = (const float*)d_in[0];
  const float* kv      = (const float*)d_in[1];
  const float* wq      = (const float*)d_in[2];
  const float* wkv     = (const float*)d_in[3];
  const float* wo      = (const float*)d_in[4];
  const float* q_norm  = (const float*)d_in[5];
  const float* k_norm  = (const float*)d_in[6];
  const int*   pos_q   = (const int*)d_in[7];
  const int*   pos_k   = (const int*)d_in[8];
  float* out = (float*)d_out;

  // workspace carve-out (256B aligned)
  char* ws = (char*)d_ws;
  size_t off = 0;
  auto carve = [&](size_t bytes) { char* p = ws + off; off += (bytes + 255) & ~(size_t)255; return p; };
  __bf16* qbf   = (__bf16*)carve((size_t)BB * LQ * DD * 2);       // queries bf16
  __bf16* kvbf  = (__bf16*)carve((size_t)BB * LK * DD * 2);       // kv bf16
  __bf16* wqbf  = (__bf16*)carve((size_t)DD * DD * 2);
  __bf16* wkvbf = (__bf16*)carve((size_t)2 * DD * DD * 2);
  __bf16* wobf  = (__bf16*)carve((size_t)DD * DD * 2);
  float*  Qp    = (float*) carve((size_t)BB * LQ * DD * 4);       // q projection f32
  float*  KVp   = (float*) carve((size_t)BB * LK * 2 * DD * 4);   // kv projection f32
  __bf16* qhb   = (__bf16*)carve((size_t)BB * HH * LQ * DH * 2);  // q heads (normed+rope, scaled)
  __bf16* khb   = (__bf16*)carve((size_t)BB * HH * LK * DH * 2);  // k heads
  __bf16* vhbT  = (__bf16*)carve((size_t)BB * HH * DH * LK * 2);  // v heads transposed [bh][d][lk]
  __bf16* ybf   = (__bf16*)carve((size_t)BB * LQ * DD * 2);       // attention output bf16

  // 1) convert inputs + weights to bf16
  {
    int n;
    n = BB * LQ * DD;     cvt_bf16_kernel<<<(n + 255) / 256, 256, 0, stream>>>(queries, qbf, n);
    n = BB * LK * DD;     cvt_bf16_kernel<<<(n + 255) / 256, 256, 0, stream>>>(kv, kvbf, n);
    n = DD * DD;          cvt_bf16_kernel<<<(n + 255) / 256, 256, 0, stream>>>(wq, wqbf, n);
    n = 2 * DD * DD;      cvt_bf16_kernel<<<(n + 255) / 256, 256, 0, stream>>>(wkv, wkvbf, n);
    n = DD * DD;          cvt_bf16_kernel<<<(n + 255) / 256, 256, 0, stream>>>(wo, wobf, n);
  }

  // 2) Qp = queries @ wq^T   (M=4096, N=1024, K=1024)
  {
    dim3 grid(DD / 128, (BB * LQ) / 64);
    gemm_nt_bf16<<<grid, 256, 0, stream>>>(qbf, wqbf, Qp, BB * LQ, DD, DD);
  }
  // 3) KVp = kv @ wkv^T      (M=8192, N=2048, K=1024)
  {
    dim3 grid((2 * DD) / 128, (BB * LK) / 64);
    gemm_nt_bf16<<<grid, 256, 0, stream>>>(kvbf, wkvbf, KVp, BB * LK, 2 * DD, DD);
  }

  // 4) RMSNorm + RoPE (+scale) -> head-major bf16 (V stored transposed)
  norm_rope_q_kernel<<<BB * LQ * HH, DH, 0, stream>>>(Qp, q_norm, pos_q, qhb);
  norm_rope_kv_kernel<<<BB * LK * HH, DH, 0, stream>>>(KVp, k_norm, pos_k, khb, vhbT);

  // 5) attention: B*H*(LQ/16) waves, 8 waves/block
  {
    int nwaves = BB * HH * (LQ / 16);
    attn_kernel<<<nwaves / 8, 256, 0, stream>>>(qhb, khb, vhbT, ybf);
  }

  // 6) out = y @ wo^T -> f32 d_out  (M=4096, N=1024, K=1024)
  {
    dim3 grid(DD / 128, (BB * LQ) / 64);
    gemm_nt_bf16<<<grid, 256, 0, stream>>>(ybf, wobf, out, BB * LQ, DD, DD);
  }
}
